// SparseMixtureOfExperts_51032801411478
// MI455X (gfx1250) — matrix operand order repaired and checked by reading, not compiled, not run on hardware
//
#include <hip/hip_runtime.h>

#define N_TOK  8192
#define EMB    1024
#define HID    4096
#define NEXP   8

typedef __attribute__((ext_vector_type(16))) __bf16       v16bf;
typedef __attribute__((ext_vector_type(8)))  float        v8f;
typedef __attribute__((ext_vector_type(8)))  unsigned int v8u;
typedef __attribute__((ext_vector_type(2)))  int          v2i;
typedef __attribute__((ext_vector_type(2)))  float        v2f;
typedef __attribute__((ext_vector_type(2)))  __bf16       v2bf;

typedef __attribute__((address_space(1))) v2i gv2i;   // global v2i
typedef __attribute__((address_space(3))) v2i lv2i;   // LDS v2i

#if __has_builtin(__builtin_amdgcn_global_load_async_to_lds_b64)
#define ASYNC_BUILTIN 1
#endif

__device__ __forceinline__ void wait_async0() {
#if __has_builtin(__builtin_amdgcn_s_wait_asynccnt)
  __builtin_amdgcn_s_wait_asynccnt(0);
#else
  asm volatile("s_wait_asynccnt 0x0" ::: "memory");
#endif
}

__device__ __forceinline__ void stage_a_async(const unsigned int* src,
                                              unsigned int* dst) {
#if ASYNC_BUILTIN
  __builtin_amdgcn_global_load_async_to_lds_b64((gv2i*)src, (lv2i*)dst, 0, 0);
#else
  unsigned int ldsoff =
      (unsigned int)(unsigned long)(__attribute__((address_space(3))) unsigned int*)dst;
  asm volatile("global_load_async_to_lds_b64 %0, %1, off"
               :: "v"(ldsoff), "v"((unsigned long long)src)
               : "memory");
#endif
}

// Packed bf16 convert (lo = a, hi = b) -> single v_cvt_pk_bf16_f32.
__device__ __forceinline__ unsigned int pack_bf16(float a, float b) {
  v2f t; t[0] = a; t[1] = b;
  v2bf r = __builtin_convertvector(t, v2bf);
  return __builtin_bit_cast(unsigned int, r);
}
__device__ __forceinline__ float bf2f(unsigned short h) {
  unsigned int u = ((unsigned int)h) << 16;
  return __builtin_bit_cast(float, u);
}

// Pack 16 f32 k-values of two columns into the [col][kpair] LDS layout.
__device__ __forceinline__ void pack_store_b(const float2* v, unsigned int* d0) {
  uint4 c0a, c0b, c1a, c1b;
  c0a.x = pack_bf16(v[0].x,  v[1].x);  c0a.y = pack_bf16(v[2].x,  v[3].x);
  c0a.z = pack_bf16(v[4].x,  v[5].x);  c0a.w = pack_bf16(v[6].x,  v[7].x);
  c0b.x = pack_bf16(v[8].x,  v[9].x);  c0b.y = pack_bf16(v[10].x, v[11].x);
  c0b.z = pack_bf16(v[12].x, v[13].x); c0b.w = pack_bf16(v[14].x, v[15].x);
  c1a.x = pack_bf16(v[0].y,  v[1].y);  c1a.y = pack_bf16(v[2].y,  v[3].y);
  c1a.z = pack_bf16(v[4].y,  v[5].y);  c1a.w = pack_bf16(v[6].y,  v[7].y);
  c1b.x = pack_bf16(v[8].y,  v[9].y);  c1b.y = pack_bf16(v[10].y, v[11].y);
  c1b.z = pack_bf16(v[12].y, v[13].y); c1b.w = pack_bf16(v[14].y, v[15].y);
  unsigned int* d1 = d0 + 20;
  ((uint4*)d0)[0] = c0a;  ((uint4*)(d0 + 4))[0] = c0b;
  ((uint4*)d1)[0] = c1a;  ((uint4*)(d1 + 4))[0] = c1b;
}

// ---------------------------------------------------------------------------
// WMMA GEMM tile: C[32 x 256] = bf16A[32 x K] * f32B[K x 256] + bias
// 256 threads = 8 waves; wave (wm,wn) owns 16x64 = 4x v8f accumulators.
// Double-buffered LDS; per iteration: issue next global B loads + next async
// A-to-LDS copy, run WMMAs from current buffers while loads are in flight,
// then pack/store next B tile. s_wait_asynccnt + barrier publish buffers.
// ---------------------------------------------------------------------------
template <bool RELU, bool OUT_BF16, bool GATHER_A, bool SCATTER_OUT>
__device__ __forceinline__ void gemm_tile(
    const unsigned short* __restrict__ A,   // bf16, row stride K
    const float* __restrict__ B,            // f32 [K x N]
    const float* __restrict__ bias,         // f32 [N]
    void* __restrict__ out,                 // bf16 or f32, row stride N
    const int* __restrict__ rowMap,         // token permutation (gather/scatter)
    int row0, int valid, int rowClamp,
    int n0, int N, int K)
{
  __shared__ __align__(16) unsigned int ldsA[2][32 * 20];   // 2 x 2.5 KB
  __shared__ __align__(16) unsigned int ldsB[2][256 * 20];  // 2 x 20 KB

  const int tid  = threadIdx.x;
  const int lane = tid & 31;
  const int wave = tid >> 5;
  const int wm   = wave & 1;     // M half
  const int wn   = wave >> 1;    // N quarter
  const int ml   = lane & 15;
  const int kh   = lane >> 4;    // k-half per WMMA 16-bit operand layout

  // Accumulators initialized with bias (broadcast along M).
  v8f acc[4];
#pragma unroll
  for (int s = 0; s < 4; ++s) {
    float bv = bias[n0 + wn * 64 + s * 16 + ml];
#pragma unroll
    for (int r = 0; r < 8; ++r) acc[s][r] = bv;
  }

  // A staging: thread -> (row ar, 8-byte chunk ac) of 32x32 bf16 tile.
  const int ar = tid >> 3;
  const int ac = tid & 7;
  int asrc = row0 + ar;
  if (asrc > rowClamp) asrc = rowClamp;
  if (GATHER_A) asrc = rowMap[asrc];
  const unsigned int* __restrict__ Arow =
      (const unsigned int*)(A + (size_t)asrc * (size_t)K);
  const int aoff = ar * 20 + ac * 2;        // LDS dword offset within buffer

  // B staging: thread -> 2 columns (bc, bc+1), 16 k-rows starting at ks.
  const int bc = (tid & 127) * 2;           // 0..254
  const int ks = (tid >> 7) * 16;           // 0 or 16
  const int boff = bc * 20 + (ks >> 1);     // LDS dword offset within buffer
  const float* __restrict__ Bcol = B + (size_t)ks * (size_t)N + n0 + bc;

  // ---- prologue: stage k0 = 0 into buffer 0 ----
  stage_a_async(Arow + ac * 2, &ldsA[0][aoff]);
  {
    const float* p = Bcol;
    float2 vg[16];
#pragma unroll
    for (int k = 0; k < 16; ++k) vg[k] = *(const float2*)(p + (size_t)k * N);
    pack_store_b(vg, &ldsB[0][boff]);
  }
  wait_async0();

  const int nsteps = K >> 5;
  for (int i = 0; i < nsteps; ++i) {
    const int cur = i & 1;
    const int nxt = cur ^ 1;
    const int k0  = i << 5;
    __syncthreads();   // buf[cur] visible to all; buf[nxt] free for restage

    // ---- issue next-iteration staging (block-uniform guard) ----
    const bool more = (i + 1) < nsteps;
    float2 vg[16];
    if (more) {
      const float* p = Bcol + (size_t)(k0 + 32) * (size_t)N;
      __builtin_prefetch(p + 32 * (size_t)N, 0, 0);   // two K-steps ahead
#pragma unroll
      for (int k = 0; k < 16; ++k) vg[k] = *(const float2*)(p + (size_t)k * N);
      stage_a_async(Arow + ac * 2 + ((k0 + 32) >> 1), &ldsA[nxt][aoff]);
    }

    // ---- compute from buf[cur] while next loads are in flight ----
    v8u aw;
    {
      const unsigned int* ab = &ldsA[cur][(wm * 16 + ml) * 20 + kh * 4];
      uint4 lo = *(const uint4*)(ab);
      uint4 hi = *(const uint4*)(ab + 8);
      aw[0] = lo.x; aw[1] = lo.y; aw[2] = lo.z; aw[3] = lo.w;
      aw[4] = hi.x; aw[5] = hi.y; aw[6] = hi.z; aw[7] = hi.w;
    }
    v16bf afrag = __builtin_bit_cast(v16bf, aw);

#pragma unroll
    for (int s = 0; s < 4; ++s) {
      const int n = wn * 64 + s * 16 + ml;
      const unsigned int* bb = &ldsB[cur][n * 20 + kh * 4];
      uint4 lo = *(const uint4*)(bb);
      uint4 hi = *(const uint4*)(bb + 8);
      v8u bw;
      bw[0] = lo.x; bw[1] = lo.y; bw[2] = lo.z; bw[3] = lo.w;
      bw[4] = hi.x; bw[5] = hi.y; bw[6] = hi.z; bw[7] = hi.w;
      v16bf bfrag = __builtin_bit_cast(v16bf, bw);
      acc[s] = __builtin_amdgcn_wmma_f32_16x16x32_bf16(
          false, afrag, false, bfrag, (short)0, acc[s], false, false);
    }

    // ---- pack and publish next B tile ----
    if (more) pack_store_b(vg, &ldsB[nxt][boff]);
    wait_async0();
  }

  // ---- epilogue: ReLU (v_max) + hw bf16 convert + (scatter) store ----
#pragma unroll
  for (int s = 0; s < 4; ++s) {
    const int ng = n0 + wn * 64 + s * 16 + ml;
#pragma unroll
    for (int r = 0; r < 8; ++r) {
      int mloc = wm * 16 + kh * 8 + r;
      if (mloc >= valid) continue;
      float v = acc[s][r];
      if (RELU) v = fmaxf(v, 0.f);
      int orow = row0 + mloc;
      if (SCATTER_OUT) orow = rowMap[orow];
      if (OUT_BF16)
        ((__bf16*)out)[(size_t)orow * (size_t)N + ng] = (__bf16)v;
      else
        ((float*)out)[(size_t)orow * (size_t)N + ng] = v;
    }
  }
}

// ---------------------------------------------------------------------------
// misc layout (int*): [0..7]=counts  [8..15]=token offsets  [16..23]=cursors
//                     [24..32]=cumulative M-tile offsets (tileOff[8]=total)
// ---------------------------------------------------------------------------

__global__ void __launch_bounds__(256)
cvt_x_kernel(const float* __restrict__ x, unsigned short* __restrict__ xb,
             int* __restrict__ misc) {
  if (blockIdx.x == 0 && threadIdx.x < NEXP) misc[threadIdx.x] = 0;  // counts
  int i = (blockIdx.x * 256 + threadIdx.x) * 4;
  float4 v = *(const float4*)(x + i);
  uint2 d;
  d.x = pack_bf16(v.x, v.y);
  d.y = pack_bf16(v.z, v.w);
  *(uint2*)(xb + i) = d;
}

__global__ void __launch_bounds__(256)
router1_kernel(const unsigned short* __restrict__ xb, const float* __restrict__ Wr1,
               const float* __restrict__ br1, unsigned short* __restrict__ hb) {
  gemm_tile<true, true, false, false>(xb, Wr1, br1, hb, nullptr,
                                      blockIdx.y * 32, 32, N_TOK - 1,
                                      blockIdx.x * 256, EMB, EMB);
}

__global__ void __launch_bounds__(256)
router2_kernel(const unsigned short* __restrict__ hb, const float* __restrict__ Wr2,
               const float* __restrict__ br2, int* __restrict__ chosen,
               int* __restrict__ misc) {
  const int wave = threadIdx.x >> 5;
  const int lane = threadIdx.x & 31;
  const int tok  = blockIdx.x * 8 + wave;
  const unsigned short* hrow = hb + (size_t)tok * EMB;

  float accv[8];
#pragma unroll
  for (int e = 0; e < 8; ++e) accv[e] = 0.f;

  for (int k = lane; k < EMB; k += 32) {
    float hv = bf2f(hrow[k]);
    const float4* wp = (const float4*)(Wr2 + (size_t)k * 8);
    float4 w0 = wp[0], w1 = wp[1];
    accv[0] += hv * w0.x; accv[1] += hv * w0.y;
    accv[2] += hv * w0.z; accv[3] += hv * w0.w;
    accv[4] += hv * w1.x; accv[5] += hv * w1.y;
    accv[6] += hv * w1.z; accv[7] += hv * w1.w;
  }
#pragma unroll
  for (int off = 16; off > 0; off >>= 1)
#pragma unroll
    for (int e = 0; e < 8; ++e) accv[e] += __shfl_xor(accv[e], off, 32);

  if (lane == 0) {
    int best = 0;
    float bv = accv[0] + br2[0];
#pragma unroll
    for (int e = 1; e < 8; ++e) {
      float v = accv[e] + br2[e];
      if (v > bv) { bv = v; best = e; }   // first-max wins, matches argmax
    }
    chosen[tok] = best;
    atomicAdd(&misc[best], 1);
  }
}

__global__ void prefix_kernel(int* __restrict__ misc) {
  if (threadIdx.x == 0 && blockIdx.x == 0) {
    int* counts  = misc;
    int* offs    = misc + 8;
    int* cursor  = misc + 16;
    int* tileOff = misc + 24;
    int acc = 0, tacc = 0;
    for (int e = 0; e < NEXP; ++e) {
      offs[e]    = acc;
      cursor[e]  = acc;
      tileOff[e] = tacc;
      acc  += counts[e];
      tacc += (counts[e] + 31) >> 5;
    }
    tileOff[NEXP] = tacc;
  }
}

__global__ void __launch_bounds__(256)
scatter_kernel(const int* __restrict__ chosen, int* __restrict__ misc,
               int* __restrict__ perm) {
  int i = blockIdx.x * 256 + threadIdx.x;
  int e = chosen[i];
  int pos = atomicAdd(&misc[16 + e], 1);
  perm[pos] = i;
}

__device__ __forceinline__ bool resolve_tile(const int* __restrict__ misc, int t,
                                             int& e, int& row0, int& valid) {
  const int* counts  = misc;
  const int* offs    = misc + 8;
  const int* tileOff = misc + 24;
  if (t >= tileOff[NEXP]) return false;
  e = 0;
  while (e < NEXP - 1 && t >= tileOff[e + 1]) e++;
  int lt = t - tileOff[e];
  row0 = offs[e] + lt * 32;
  int rem = counts[e] - lt * 32;
  valid = rem < 32 ? rem : 32;
  return true;
}

__global__ void __launch_bounds__(256)
ffn1_kernel(const unsigned short* __restrict__ xb, const float* __restrict__ W1,
            const float* __restrict__ b1, unsigned short* __restrict__ hidb,
            const int* __restrict__ perm, const int* __restrict__ misc) {
  int e, row0, valid;
  if (!resolve_tile(misc, blockIdx.y, e, row0, valid)) return;
  gemm_tile<true, true, true, false>(
      xb, W1 + (size_t)e * EMB * HID, b1 + (size_t)e * HID, hidb, perm,
      row0, valid, N_TOK - 1, blockIdx.x * 256, HID, EMB);
}

__global__ void __launch_bounds__(256)
ffn2_kernel(const unsigned short* __restrict__ hidb, const float* __restrict__ W2,
            const float* __restrict__ b2, float* __restrict__ out,
            const int* __restrict__ perm, const int* __restrict__ misc) {
  int e, row0, valid;
  if (!resolve_tile(misc, blockIdx.y, e, row0, valid)) return;
  gemm_tile<false, false, false, true>(
      hidb, W2 + (size_t)e * HID * EMB, b2 + (size_t)e * EMB, out, perm,
      row0, valid, N_TOK - 1, blockIdx.x * 256, EMB, HID);
}

// ---------------------------------------------------------------------------
extern "C" void kernel_launch(void* const* d_in, const int* in_sizes, int n_in,
                              void* d_out, int out_size, void* d_ws, size_t ws_size,
                              hipStream_t stream) {
  (void)in_sizes; (void)n_in; (void)out_size; (void)ws_size;

  const float* x   = (const float*)d_in[0];
  const float* Wr1 = (const float*)d_in[1];
  const float* br1 = (const float*)d_in[2];
  const float* Wr2 = (const float*)d_in[3];
  const float* br2 = (const float*)d_in[4];
  const float* W1  = (const float*)d_in[5];
  const float* b1  = (const float*)d_in[6];
  const float* W2  = (const float*)d_in[7];
  const float* b2  = (const float*)d_in[8];
  float* out = (float*)d_out;

  char* ws = (char*)d_ws;
  int* chosen = (int*)(ws);                 // 8192 ints
  int* perm   = (int*)(ws + 32768);         // 8192 ints
  int* misc   = (int*)(ws + 65536);         // counts/offs/cursor/tileOff
  unsigned short* xb   = (unsigned short*)(ws + 66560);        // 16 MB bf16 x
  unsigned short* hb   = xb + (size_t)N_TOK * EMB;             // 16 MB router hidden
  unsigned short* hidb = hb + (size_t)N_TOK * EMB;             // 64 MB FFN hidden

  const int MAX_TILES = N_TOK / 32 + NEXP;  // 264 worst-case M tiles

  cvt_x_kernel  <<<dim3(N_TOK * EMB / 1024), dim3(256), 0, stream>>>(x, xb, misc);
  router1_kernel<<<dim3(EMB / 256, N_TOK / 32), dim3(256), 0, stream>>>(xb, Wr1, br1, hb);
  router2_kernel<<<dim3(N_TOK / 8), dim3(256), 0, stream>>>(hb, Wr2, br2, chosen, misc);
  prefix_kernel <<<dim3(1), dim3(32), 0, stream>>>(misc);
  scatter_kernel<<<dim3(N_TOK / 256), dim3(256), 0, stream>>>(chosen, misc, perm);
  ffn1_kernel   <<<dim3(HID / 256, MAX_TILES), dim3(256), 0, stream>>>(xb, W1, b1, hidb, perm, misc);
  ffn2_kernel   <<<dim3(EMB / 256, MAX_TILES), dim3(256), 0, stream>>>(hidb, W2, b2, out, perm, misc);
}
